// CoWindowAttention_249108103598
// MI455X (gfx1250) — matrix operand order repaired and checked by reading, not compiled, not run on hardware
//
#include <hip/hip_runtime.h>

typedef __attribute__((ext_vector_type(16))) _Float16 v16h;
typedef __attribute__((ext_vector_type(8)))  _Float16 v8h;
typedef __attribute__((ext_vector_type(8)))  float    v8f;

// ---------------------------------------------------------------------------
// Fragment loaders for V_WMMA_F32_16X16X32_F16 (wave32).
// A-matrix 16x32 f16 (ISA 7.12.2): lanes 0-15 hold row M=lane, K={0..7,16..23};
// lanes 16-31 hold row M=lane-16, K={8..15,24..31}.  -> hi8 = (lane&16)?8:0
// B-matrix 32x16 f16 (same K order as A): lanes 0-15 hold col N=lane,
// K={0..7,16..23}; lanes 16-31 hold col N=lane-16, K={8..15,24..31}. -> hi16>>1
// ---------------------------------------------------------------------------
__device__ __forceinline__ v16h load_a_frag(const _Float16* p, int stride,
                                            int row0, int k0, int lo, int hi8) {
  const _Float16* q = p + (row0 + lo) * stride + k0 + hi8;
  v8h a0 = *(const v8h*)(q);
  v8h a1 = *(const v8h*)(q + 16);
  return __builtin_shufflevector(a0, a1, 0,1,2,3,4,5,6,7,8,9,10,11,12,13,14,15);
}

__device__ __forceinline__ v16h load_b_frag(const _Float16* p, int stride,
                                            int col0, int k0, int lo, int hi16) {
  const _Float16* q = p + (col0 + lo) * stride + k0 + (hi16 >> 1);
  v8h a0 = *(const v8h*)(q);
  v8h a1 = *(const v8h*)(q + 16);
  return __builtin_shufflevector(a0, a1, 0,1,2,3,4,5,6,7,8,9,10,11,12,13,14,15);
}

#define WMMA_F16(A, B, C) \
  __builtin_amdgcn_wmma_f32_16x16x32_f16(false, (A), false, (B), (short)0, (C), false, false)

// ---------------------------------------------------------------------------
// Convert the four 512x512 f32 weight matrices to f16 in workspace.
// ---------------------------------------------------------------------------
__global__ void cvt_weights(const float* __restrict__ a, const float* __restrict__ b,
                            const float* __restrict__ c, const float* __restrict__ d,
                            _Float16* __restrict__ o) {
  int i = blockIdx.x * blockDim.x + threadIdx.x;
  if (i < 262144) {
    o[i]           = (_Float16)a[i];
    o[262144 + i]  = (_Float16)b[i];
    o[524288 + i]  = (_Float16)c[i];
    o[786432 + i]  = (_Float16)d[i];
  }
}

// ---------------------------------------------------------------------------
// Branchless depthwise 3x3 + bias + relu -> sR (f16), one window.
// Clamped (always-valid) address + cndmask'd weight instead of branches.
// ---------------------------------------------------------------------------
__device__ __forceinline__ void dwconv_relu(const float* __restrict__ xb,
                                            const float* __restrict__ dw,
                                            const float* __restrict__ db,
                                            _Float16* sR, int tid) {
  for (int e = tid; e < 64 * 512; e += 256) {
    int n = e >> 9, c = e & 511;
    int i = n >> 3, j = n & 7;
    float s = db[c];
#pragma unroll
    for (int di = 0; di < 3; ++di) {
      int ii = i + di - 1;
      bool vi = (unsigned)ii < 8u;
      int iic = vi ? ii : i;                       // always-valid row
#pragma unroll
      for (int dj = 0; dj < 3; ++dj) {
        int jj = j + dj - 1;
        bool vv = vi && ((unsigned)jj < 8u);
        int jjc = vv ? jj : j;                     // always-valid col
        float wgt = dw[c * 9 + di * 3 + dj];       // always-valid address
        wgt = vv ? wgt : 0.f;                      // cndmask, no branch
        s += xb[(iic * 8 + jjc) * 512 + c] * wgt;
      }
    }
    sR[e] = (_Float16)fmaxf(s, 0.f);
  }
}

// ---------------------------------------------------------------------------
// Pointwise GEMM [64x512] = sSrc[64x512] x W^T, specialized epilogue:
//   MODE 0: scale by 1/sqrt(32), store row-major to dstN   (q)
//   MODE 1: store row-major to dstN                        (k)
//   MODE 2: store transposed [512x64] to dstT              (v)
// W passed directly (no runtime pointer select) so loads stay global_load.
// ---------------------------------------------------------------------------
template <int MODE>
__device__ __forceinline__ void projection_gemm(const _Float16* __restrict__ W,
                                                const float* __restrict__ pb,
                                                const _Float16* sSrc,
                                                _Float16* dstN, _Float16* dstT,
                                                int wid, int lo, int hi8, int hi16) {
  const int colbase = wid * 64;
  const v8f zero = {0.f, 0.f, 0.f, 0.f, 0.f, 0.f, 0.f, 0.f};
  v8f acc[16];
#pragma unroll
  for (int t = 0; t < 16; ++t) acc[t] = zero;
  for (int kc = 0; kc < 16; ++kc) {
    const int k0 = kc * 32;
    v16h A[4];
#pragma unroll
    for (int rt = 0; rt < 4; ++rt) A[rt] = load_a_frag(sSrc, 512, rt * 16, k0, lo, hi8);
#pragma unroll
    for (int ct = 0; ct < 4; ++ct) {
      v16h Bf = load_b_frag(W, 512, colbase + ct * 16, k0, lo, hi16);
#pragma unroll
      for (int rt = 0; rt < 4; ++rt)
        acc[rt * 4 + ct] = WMMA_F16(A[rt], Bf, acc[rt * 4 + ct]);
    }
  }
  const float scl = (MODE == 0) ? 0.17677669529663687f : 1.0f;  // (DIM/HEADS)^-0.5
#pragma unroll
  for (int rt = 0; rt < 4; ++rt)
#pragma unroll
    for (int ct = 0; ct < 4; ++ct) {
      int N = colbase + ct * 16 + lo;
      float bias = pb[N];
#pragma unroll
      for (int r = 0; r < 8; ++r) {
        int M = rt * 16 + r + hi8;
        float val = (acc[rt * 4 + ct][r] + bias) * scl;
        if (MODE == 2) dstT[N * 64 + M] = (_Float16)val;   // compile-time branch
        else           dstN[M * 512 + N] = (_Float16)val;
      }
    }
}

// ---------------------------------------------------------------------------
// Fused window attention: one block per window (b), 256 threads = 8 waves.
// ---------------------------------------------------------------------------
__global__ __launch_bounds__(256) void cowin_attn(
    const float* __restrict__ x, const float* __restrict__ mask,
    const float* __restrict__ dwq_w, const float* __restrict__ dwq_b,
    const _Float16* __restrict__ wq, const float* __restrict__ pbq,
    const float* __restrict__ dwk_w, const float* __restrict__ dwk_b,
    const _Float16* __restrict__ wk, const float* __restrict__ pbk,
    const float* __restrict__ dwv_w, const float* __restrict__ dwv_b,
    const _Float16* __restrict__ wv, const float* __restrict__ pbv,
    const float* __restrict__ pos, const _Float16* __restrict__ wproj,
    const float* __restrict__ pjb, float* __restrict__ out) {

  __shared__ _Float16 sQ[64 * 512]   __attribute__((aligned(16)));   // 64 KB (q, pre-scaled)
  __shared__ _Float16 sK[64 * 512]   __attribute__((aligned(16)));   // 64 KB
  __shared__ _Float16 sVT[512 * 64]  __attribute__((aligned(16)));   // 64 KB (v transposed)
  __shared__ _Float16 sR[64 * 512]   __attribute__((aligned(16)));   // 64 KB (dwconv y, then attn-out)
  __shared__ _Float16 sStage[8*1024] __attribute__((aligned(16)));   // 16 KB (per-wave attn probs)

  const int b    = blockIdx.x;
  const int tid  = threadIdx.x;
  const int wid  = tid >> 5;
  const int lane = tid & 31;
  const int lo   = lane & 15;
  const int hi8  = (lane & 16) ? 8 : 0;
  const int hi16 = (lane & 16) ? 16 : 0;

  const float* xb = x + (size_t)b * 32768;
  const v8f zero = {0.f, 0.f, 0.f, 0.f, 0.f, 0.f, 0.f, 0.f};

  // ============ Phase 1: q/k/v = pointwise( relu( depthwise(x) ) ) ============
  dwconv_relu(xb, dwq_w, dwq_b, sR, tid);
  __syncthreads();
  projection_gemm<0>(wq, pbq, sR, sQ, nullptr, wid, lo, hi8, hi16);
  __syncthreads();

  dwconv_relu(xb, dwk_w, dwk_b, sR, tid);
  __syncthreads();
  projection_gemm<1>(wk, pbk, sR, sK, nullptr, wid, lo, hi8, hi16);
  __syncthreads();

  dwconv_relu(xb, dwv_w, dwv_b, sR, tid);
  __syncthreads();
  projection_gemm<2>(wv, pbv, sR, nullptr, sVT, wid, lo, hi8, hi16);
  __syncthreads();

  // ============ Phase 2: attention (each wave handles 2 heads) ============
  const float* mk = mask + (size_t)b * 4096;
  for (int hh = 0; hh < 2; ++hh) {
    const int h = wid * 2 + hh;
    v8f acc[16];
#pragma unroll
    for (int t = 0; t < 16; ++t) acc[t] = zero;
    // attn = (q*scale) @ k^T   (K = head_dim = 32, single k-chunk)
    {
      v16h A[4];
#pragma unroll
      for (int rt = 0; rt < 4; ++rt) A[rt] = load_a_frag(sQ, 512, rt * 16, h * 32, lo, hi8);
#pragma unroll
      for (int ct = 0; ct < 4; ++ct) {
        v16h Bf = load_b_frag(sK + h * 32, 512, ct * 16, 0, lo, hi16);
#pragma unroll
        for (int rt = 0; rt < 4; ++rt)
          acc[rt * 4 + ct] = WMMA_F16(A[rt], Bf, acc[rt * 4 + ct]);
      }
    }
    // + pos_emb + shift mask
#pragma unroll
    for (int rt = 0; rt < 4; ++rt)
#pragma unroll
      for (int ct = 0; ct < 4; ++ct) {
        int m0 = ct * 16 + lo;
#pragma unroll
        for (int r = 0; r < 8; ++r) {
          int n = rt * 16 + r + hi8;
          acc[rt * 4 + ct][r] += pos[n * 64 + m0] + mk[n * 64 + m0];
        }
      }
    // per 16-row tile: softmax -> stage f16 -> attn @ v
    _Float16* stg = sStage + wid * 1024;
    const _Float16* vh = sVT + h * 32 * 64;
#pragma unroll
    for (int tm = 0; tm < 4; ++tm) {
#pragma unroll
      for (int r = 0; r < 8; ++r) {
        float a0 = acc[tm * 4 + 0][r], a1 = acc[tm * 4 + 1][r];
        float a2 = acc[tm * 4 + 2][r], a3 = acc[tm * 4 + 3][r];
        float mx = fmaxf(fmaxf(a0, a1), fmaxf(a2, a3));
        mx = fmaxf(mx, __shfl_xor(mx, 1, 32));
        mx = fmaxf(mx, __shfl_xor(mx, 2, 32));
        mx = fmaxf(mx, __shfl_xor(mx, 4, 32));
        mx = fmaxf(mx, __shfl_xor(mx, 8, 32));   // row lives in a 16-lane group
        float e0 = __expf(a0 - mx), e1 = __expf(a1 - mx);
        float e2 = __expf(a2 - mx), e3 = __expf(a3 - mx);
        float sm = e0 + e1 + e2 + e3;
        sm += __shfl_xor(sm, 1, 32);
        sm += __shfl_xor(sm, 2, 32);
        sm += __shfl_xor(sm, 4, 32);
        sm += __shfl_xor(sm, 8, 32);
        float inv = 1.0f / sm;
        int rr = r + hi8;
        _Float16* sp = stg + rr * 64 + lo;
        sp[0]  = (_Float16)(e0 * inv);
        sp[16] = (_Float16)(e1 * inv);
        sp[32] = (_Float16)(e2 * inv);
        sp[48] = (_Float16)(e3 * inv);
      }
      asm volatile("s_wait_dscnt 0" ::: "memory");   // per-wave LDS RAW fence
      v8f o0 = zero, o1 = zero;
#pragma unroll
      for (int kc = 0; kc < 2; ++kc) {
        v16h Aa = load_a_frag(stg, 64, 0, kc * 32, lo, hi8);
        v16h B0 = load_b_frag(vh, 64, 0,  kc * 32, lo, hi16);
        v16h B1 = load_b_frag(vh, 64, 16, kc * 32, lo, hi16);
        o0 = WMMA_F16(Aa, B0, o0);
        o1 = WMMA_F16(Aa, B1, o1);
      }
#pragma unroll
      for (int r = 0; r < 8; ++r) {
        int M = tm * 16 + r + hi8;
        sR[M * 512 + h * 32 + lo]      = (_Float16)o0[r];
        sR[M * 512 + h * 32 + 16 + lo] = (_Float16)o1[r];
      }
    }
  }
  __syncthreads();

  // ============ Phase 3: out = attn_out @ proj_w^T + proj_b ============
  {
    const int colbase = wid * 64;
    v8f acc[16];
#pragma unroll
    for (int t = 0; t < 16; ++t) acc[t] = zero;
    for (int kc = 0; kc < 16; ++kc) {
      const int k0 = kc * 32;
      v16h A[4];
#pragma unroll
      for (int rt = 0; rt < 4; ++rt) A[rt] = load_a_frag(sR, 512, rt * 16, k0, lo, hi8);
#pragma unroll
      for (int ct = 0; ct < 4; ++ct) {
        v16h Bf = load_b_frag(wproj, 512, colbase + ct * 16, k0, lo, hi16);
#pragma unroll
        for (int rt = 0; rt < 4; ++rt)
          acc[rt * 4 + ct] = WMMA_F16(A[rt], Bf, acc[rt * 4 + ct]);
      }
    }
    float* ob = out + (size_t)b * 32768;
#pragma unroll
    for (int rt = 0; rt < 4; ++rt)
#pragma unroll
      for (int ct = 0; ct < 4; ++ct) {
        int N = colbase + ct * 16 + lo;
        float bias = pjb[N];
#pragma unroll
        for (int r = 0; r < 8; ++r) {
          int M = rt * 16 + r + hi8;
          ob[M * 512 + N] = acc[rt * 4 + ct][r] + bias;
        }
      }
  }
}

extern "C" void kernel_launch(void* const* d_in, const int* in_sizes, int n_in,
                              void* d_out, int out_size, void* d_ws, size_t ws_size,
                              hipStream_t stream) {
  const float* x      = (const float*)d_in[0];
  const float* mask   = (const float*)d_in[1];
  const float* dwq_w  = (const float*)d_in[2];
  const float* dwq_b  = (const float*)d_in[3];
  const float* pwq_w  = (const float*)d_in[4];
  const float* pwq_b  = (const float*)d_in[5];
  const float* dwk_w  = (const float*)d_in[6];
  const float* dwk_b  = (const float*)d_in[7];
  const float* pwk_w  = (const float*)d_in[8];
  const float* pwk_b  = (const float*)d_in[9];
  const float* dwv_w  = (const float*)d_in[10];
  const float* dwv_b  = (const float*)d_in[11];
  const float* pwv_w  = (const float*)d_in[12];
  const float* pwv_b  = (const float*)d_in[13];
  const float* pos    = (const float*)d_in[14];
  const float* proj_w = (const float*)d_in[15];
  const float* proj_b = (const float*)d_in[16];

  _Float16* wsh = (_Float16*)d_ws;   // 4 x 512x512 f16 = 2 MB
  cvt_weights<<<dim3(1024), dim3(256), 0, stream>>>(pwq_w, pwk_w, pwv_w, proj_w, wsh);

  cowin_attn<<<dim3(1024), dim3(256), 0, stream>>>(
      x, mask,
      dwq_w, dwq_b, wsh,          pwq_b,
      dwk_w, dwk_b, wsh + 262144, pwk_b,
      dwv_w, dwv_b, wsh + 524288, pwv_b,
      pos, wsh + 786432, proj_b,
      (float*)d_out);
}
